// GluedSlidingTiledFlexAttnProcessor_52553219834382
// MI455X (gfx1250) — compile-verified, matrix operand order checked by reading, not run on hardware
//
#include <hip/hip_runtime.h>
#include <cstdint>

typedef __attribute__((ext_vector_type(16))) _Float16 v16h;
typedef __attribute__((ext_vector_type(8)))  _Float16 v8h;
typedef __attribute__((ext_vector_type(8)))  float    v8f;

#define HIDDEN 2048
#define LSEQ   4096
#define HEADS  16
#define HD     128
#define NTILES 16
#define TTOK   256

// ---------------------------------------------------------------------------
// WMMA fragment load (f16, 16x32 A / 32x16 B): lane m holds row m; K elems
// 0..7 & 16..23 for lanes 0-15, 8..15 & 24..31 for lanes 16-31 (ISA 7.12.2).
// base must already point at (row0, k0) of the fragment; ld = row stride.
// Works for both global and LDS (addrspace-known) pointers.
// ---------------------------------------------------------------------------
__device__ __forceinline__ v16h frag_ld(const _Float16* base, int ld) {
  int lane = threadIdx.x & 31;
  const _Float16* p = base + (size_t)(lane & 15) * ld + ((lane >> 4) << 3);
  v8h lo = *(const v8h*)p;
  v8h hi = *(const v8h*)(p + 16);
  v16h out;
#pragma unroll
  for (int i = 0; i < 8; ++i) { out[i] = lo[i]; out[i + 8] = hi[i]; }
  return out;
}

__device__ __forceinline__ v8f wmma16(v16h a, v16h b, v8f c) {
  return __builtin_amdgcn_wmma_f32_16x16x32_f16(false, a, false, b, (short)0, c, false, false);
}

// Async copy of one 16-byte chunk global -> LDS (ASYNCcnt-tracked, GV mode).
__device__ __forceinline__ void async_cp16(uint32_t lds_off, const _Float16* gsrc) {
  asm volatile("global_load_async_to_lds_b128 %0, %1, off"
               :: "v"(lds_off), "v"(gsrc) : "memory");
}
__device__ __forceinline__ void wait_async_0() {
  asm volatile("s_wait_asynccnt 0x0" ::: "memory");
}
__device__ __forceinline__ void wait_async_16() {
  asm volatile("s_wait_asynccnt 0x10" ::: "memory");
}

// ---------------------------------------------------------------------------
// Kernel 1a: hidden fp32 -> f16
// ---------------------------------------------------------------------------
__global__ __launch_bounds__(256) void hcvt_kernel(const float* __restrict__ h,
                                                   _Float16* __restrict__ h16) {
  int idx = blockIdx.x * 256 + threadIdx.x;  // 8M elems
  h16[idx] = (_Float16)h[idx];
}

// Kernel 1b: weight fp32 [K][N] -> f16 transposed [N][K]
__global__ __launch_bounds__(256) void wcvt_kernel(const float* __restrict__ w,
                                                   _Float16* __restrict__ wt) {
  int idx = blockIdx.x * 256 + threadIdx.x;  // 4M elems
  int n = idx >> 11;
  int k = idx & 2047;
  wt[idx] = (_Float16)w[(size_t)k * HIDDEN + n];
}

// ---------------------------------------------------------------------------
// Kernel 2: fused QKV GEMM + bias + RMSNorm + RoPE + tiling scatter.
// grid.x = 64 row-blocks of 64, grid.y = 3 mats * 16 heads. 256 thr = 8 waves;
// wave (wM in 0..3, wN in 0..1) computes a 16x64 sub-block of the 64x128 tile.
// ---------------------------------------------------------------------------
__global__ __launch_bounds__(256) void qkv_kernel(
    const _Float16* __restrict__ h16,
    const _Float16* __restrict__ wq16, const _Float16* __restrict__ wk16,
    const _Float16* __restrict__ wv16,
    const float* __restrict__ bq, const float* __restrict__ bk,
    const float* __restrict__ bv,
    const float* __restrict__ gq, const float* __restrict__ gk,
    _Float16* __restrict__ q_t, _Float16* __restrict__ k_t,
    _Float16* __restrict__ v_t) {
  const int mat  = blockIdx.y >> 4;   // 0=q 1=k 2=v
  const int head = blockIdx.y & 15;
  const int rowBase = blockIdx.x * 64;
  const int tid = threadIdx.x, wave = tid >> 5, lane = tid & 31;
  const int wM = wave >> 1, wN = wave & 1;
  const int ln = lane & 15, hi = lane >> 4;

  const _Float16* wt = (mat == 0 ? wq16 : (mat == 1 ? wk16 : wv16)) +
                       (size_t)(head * HD) * HIDDEN;
  const float* bias = (mat == 0 ? bq : (mat == 1 ? bk : bv)) + head * HD;

  const int rowW = rowBase + wM * 16;
  v8f acc[4] = {};
#pragma unroll 4
  for (int kb = 0; kb < HIDDEN; kb += 32) {
    v16h a = frag_ld(h16 + (size_t)rowW * HIDDEN + kb, HIDDEN);
#pragma unroll
    for (int c = 0; c < 4; ++c) {
      v16h b = frag_ld(wt + (size_t)(wN * 64 + c * 16) * HIDDEN + kb, HIDDEN);
      acc[c] = wmma16(a, b, acc[c]);
    }
  }

  // per-lane column constants (e is head-local feature index 0..127)
  float bcol[4], gcol[4], freqc[4];
  int axc[4], ecol[4];
#pragma unroll
  for (int c = 0; c < 4; ++c) {
    int e = wN * 64 + c * 16 + ln;
    ecol[c] = e;
    bcol[c] = bias[e];
    gcol[c] = 1.0f;
    int base, dim, ax;
    if (e < 16)      { base = 0;  dim = 16; ax = 0; }
    else if (e < 72) { base = 16; dim = 56; ax = 1; }
    else             { base = 72; dim = 56; ax = 2; }
    int p = (e - base) >> 1;
    freqc[c] = __expf(-((float)(2 * p) / (float)dim) * 5.545177444479562f); // ln(256)
    axc[c] = ax;
    if (mat == 0) gcol[c] = gq[e];
    else if (mat == 1) gcol[c] = gk[e];
  }
#pragma unroll
  for (int c = 0; c < 4; ++c)
#pragma unroll
    for (int r = 0; r < 8; ++r) acc[c][r] += bcol[c];

  if (mat == 2) {  // V: no norm/rope; store transposed per tile [h][n][e][t]
#pragma unroll
    for (int r = 0; r < 8; ++r) {
      int l = rowW + r + hi * 8;
      int c1 = l >> 10, c2 = (l >> 5) & 31, c3 = l & 31;
      int n = ((c2 >> 3) << 2) + (c3 >> 3);
      int t = ((c1 & 3) << 6) + ((c2 & 7) << 3) + (c3 & 7);
#pragma unroll
      for (int c = 0; c < 4; ++c)
        v_t[((size_t)(head * NTILES + n) * HD + ecol[c]) * TTOK + t] =
            (_Float16)acc[c][r];
    }
    return;
  }

  // RMS: row sums of squares; partial across 64 cols, combine two waves via LDS
  __shared__ float red[4][16][2];
  float ss[8];
#pragma unroll
  for (int r = 0; r < 8; ++r) {
    float s = 0.f;
#pragma unroll
    for (int c = 0; c < 4; ++c) s += acc[c][r] * acc[c][r];
    s += __shfl_xor(s, 1, 32);
    s += __shfl_xor(s, 2, 32);
    s += __shfl_xor(s, 4, 32);
    s += __shfl_xor(s, 8, 32);
    ss[r] = s;
  }
  if (ln == 0) {
#pragma unroll
    for (int r = 0; r < 8; ++r) red[wM][r + hi * 8][wN] = ss[r];
  }
  __syncthreads();

  _Float16* dst = (mat == 0) ? q_t : k_t;
  const float qscale = (mat == 0) ? 0.08838834764831845f : 1.0f; // HD^-0.5 folded into Q
#pragma unroll
  for (int r = 0; r < 8; ++r) {
    int row8 = r + hi * 8;
    float tot = red[wM][row8][0] + red[wM][row8][1];
    float rr = rsqrtf(tot * (1.0f / 128.0f) + 1e-6f);
    int l = rowW + row8;
    int c1 = l >> 10, c2 = (l >> 5) & 31, c3 = l & 31;
    float pos3[3] = {(float)c1, (float)c2, (float)c3};
    int n = ((c2 >> 3) << 2) + (c3 >> 3);
    int t = ((c1 & 3) << 6) + ((c2 & 7) << 3) + (c3 & 7);
#pragma unroll
    for (int c = 0; c < 4; ++c) {
      float x = acc[c][r] * rr * gcol[c];
      float part = __shfl_xor(x, 1, 32);              // rotate-half partner
      float rot = (ecol[c] & 1) ? part : -part;
      float ang = pos3[axc[c]] * freqc[c];
      float sn, cs;
      __sincosf(ang, &sn, &cs);
      float y = (x * cs + rot * sn) * qscale;
      dst[((size_t)(head * NTILES + n) * TTOK + t) * HD + ecol[c]] = (_Float16)y;
    }
  }
}

// ---------------------------------------------------------------------------
// Kernel 3: windowed flash attention with double-buffered async K/V staging.
// grid = (tile n, head, row-half). 8 waves; each wave owns a 16-row Q strip,
// loops 18 half-tiles (9 kv tiles split into 128-col halves) with online
// softmax. K/V half-tiles (32KB each) are staged into LDS once per workgroup
// via global_load_async_to_lds_b128 (ASYNCcnt), prefetching step s+1 while
// computing step s. P goes through LDS (C-layout -> A-fragment layout).
// ---------------------------------------------------------------------------
__global__ __launch_bounds__(256) void attn_kernel(
    const _Float16* __restrict__ q_t, const _Float16* __restrict__ k_t,
    const _Float16* __restrict__ v_t, _Float16* __restrict__ o16) {
  const int n = blockIdx.x, head = blockIdx.y, half = blockIdx.z;
  const int tid = threadIdx.x, wave = tid >> 5, lane = tid & 31;
  const int ln = lane & 15, hi = lane >> 4;
  const int qr = half * 128 + wave * 16;

  const _Float16* qh = q_t + (size_t)(head * NTILES + n) * TTOK * HD;
  v16h qf[4];
#pragma unroll
  for (int kc = 0; kc < 4; ++kc)
    qf[kc] = frag_ld(qh + (size_t)qr * HD + kc * 32, HD);

  v8f o[8] = {};
  float mrun[8], lrun[8];
#pragma unroll
  for (int r = 0; r < 8; ++r) { mrun[r] = -1e30f; lrun[r] = 0.f; }

  const int d2 = n >> 2, d3 = n & 3;
  const int s2 = min(max(d2 - 1, 0), 1);
  const int s3 = min(max(d3 - 1, 0), 1);

  // 2x(K 128x128) + 2x(V 128x128) + P staging = 160KB LDS
  __shared__ __align__(16) _Float16 klds[2][128][128];
  __shared__ __align__(16) _Float16 vlds[2][128][128];
  __shared__ __align__(16) _Float16 plds[8][16][128];

  // issue async copies of K/V half-tile for 'step' into buffer step&1
  auto issue_prefetch = [&](int step) {
    const int kk = step >> 1, ch = step & 1, b = step & 1;
    const int kvn = (s2 + kk / 3) * 4 + (s3 + kk % 3);
    const _Float16* kh = k_t + (size_t)(head * NTILES + kvn) * TTOK * HD +
                         (size_t)(ch * 128) * HD;
    const _Float16* vh = v_t + (size_t)(head * NTILES + kvn) * HD * TTOK +
                         ch * 128;
#pragma unroll
    for (int i = 0; i < 8; ++i) {
      int chunk = tid + 256 * i;          // 2048 16B chunks per 32KB tile
      int row = chunk >> 4;
      int c8 = (chunk & 15) << 3;
      async_cp16((uint32_t)(uintptr_t)&klds[b][row][c8], kh + row * HD + c8);
      async_cp16((uint32_t)(uintptr_t)&vlds[b][row][c8], vh + row * TTOK + c8);
    }
  };

  issue_prefetch(0);

  for (int step = 0; step < 18; ++step) {
    const int buf = step & 1;
    if (step + 1 < 18) {
      issue_prefetch(step + 1);  // up to 32 outstanding per wave
      wait_async_16();           // in-order: step's 16 ops have landed
    } else {
      wait_async_0();
    }
    __syncthreads();             // every wave's copies visible to all

    // S = Q * K^T for a 16x128 strip (K fragments from LDS)
    v8f sacc[8] = {};
#pragma unroll
    for (int jc = 0; jc < 8; ++jc)
#pragma unroll
      for (int kc = 0; kc < 4; ++kc) {
        v16h b = frag_ld(&klds[buf][jc * 16][kc * 32], 128);
        sacc[jc] = wmma16(qf[kc], b, sacc[jc]);
      }

    // online softmax update per row
#pragma unroll
    for (int r = 0; r < 8; ++r) {
      float mx = sacc[0][r];
#pragma unroll
      for (int jc = 1; jc < 8; ++jc) mx = fmaxf(mx, sacc[jc][r]);
      mx = fmaxf(mx, __shfl_xor(mx, 1, 32));
      mx = fmaxf(mx, __shfl_xor(mx, 2, 32));
      mx = fmaxf(mx, __shfl_xor(mx, 4, 32));
      mx = fmaxf(mx, __shfl_xor(mx, 8, 32));
      float mnew = fmaxf(mrun[r], mx);
      float alpha = __expf(mrun[r] - mnew);
      mrun[r] = mnew;
      lrun[r] *= alpha;
      float ps = 0.f;
#pragma unroll
      for (int jc = 0; jc < 8; ++jc) {
        float p = __expf(sacc[jc][r] - mnew);
        sacc[jc][r] = p;
        ps += p;
      }
      ps += __shfl_xor(ps, 1, 32);
      ps += __shfl_xor(ps, 2, 32);
      ps += __shfl_xor(ps, 4, 32);
      ps += __shfl_xor(ps, 8, 32);
      lrun[r] += ps;
#pragma unroll
      for (int eb = 0; eb < 8; ++eb) o[eb][r] *= alpha;
    }

    // stage P (C-layout -> row-major f16 in LDS)
#pragma unroll
    for (int jc = 0; jc < 8; ++jc)
#pragma unroll
      for (int r = 0; r < 8; ++r)
        plds[wave][r + hi * 8][jc * 16 + ln] = (_Float16)sacc[jc][r];
    __syncthreads();

    // O += P * V  (V fragments from LDS, feature-major -> contiguous)
#pragma unroll
    for (int kc = 0; kc < 4; ++kc) {
      v16h a = frag_ld(&plds[wave][0][kc * 32], 128);
#pragma unroll
      for (int eb = 0; eb < 8; ++eb) {
        v16h b = frag_ld(&vlds[buf][eb * 16][kc * 32], 128);
        o[eb] = wmma16(a, b, o[eb]);
      }
    }
    __syncthreads();  // all waves done with buf before it is overwritten
  }

  // normalize + untile scatter to [L][HIDDEN] f16
#pragma unroll
  for (int r = 0; r < 8; ++r) {
    float inv = 1.0f / lrun[r];
    int t = qr + r + hi * 8;
    int a1 = t >> 6, a2 = (t >> 3) & 7, a3 = t & 7;
    int c1 = a1, c2 = d2 * 8 + a2, c3 = d3 * 8 + a3;
    int l = (c1 << 10) + (c2 << 5) + c3;
#pragma unroll
    for (int eb = 0; eb < 8; ++eb)
      o16[(size_t)l * HIDDEN + head * HD + eb * 16 + ln] =
          (_Float16)(o[eb][r] * inv);
  }
}

// ---------------------------------------------------------------------------
// Kernel 4: output projection GEMM, fp32 result + bias.
// grid = (64 row blocks, 16 col blocks of 128). Same wave tiling as kernel 2.
// ---------------------------------------------------------------------------
__global__ __launch_bounds__(256) void oproj_kernel(
    const _Float16* __restrict__ o16, const _Float16* __restrict__ wo16,
    const float* __restrict__ bo, float* __restrict__ out) {
  const int rowBase = blockIdx.x * 64;
  const int colBase = blockIdx.y * 128;
  const int tid = threadIdx.x, wave = tid >> 5, lane = tid & 31;
  const int wM = wave >> 1, wN = wave & 1;
  const int ln = lane & 15, hi = lane >> 4;
  const int rowW = rowBase + wM * 16;

  v8f acc[4] = {};
#pragma unroll 4
  for (int kb = 0; kb < HIDDEN; kb += 32) {
    v16h a = frag_ld(o16 + (size_t)rowW * HIDDEN + kb, HIDDEN);
#pragma unroll
    for (int c = 0; c < 4; ++c) {
      v16h b = frag_ld(wo16 + (size_t)(colBase + wN * 64 + c * 16) * HIDDEN + kb,
                       HIDDEN);
      acc[c] = wmma16(a, b, acc[c]);
    }
  }
#pragma unroll
  for (int c = 0; c < 4; ++c) {
    int col = colBase + wN * 64 + c * 16 + ln;
    float bb = bo[col];
#pragma unroll
    for (int r = 0; r < 8; ++r) {
      int row = rowW + r + hi * 8;
      out[(size_t)row * HIDDEN + col] = acc[c][r] + bb;
    }
  }
}

// ---------------------------------------------------------------------------
extern "C" void kernel_launch(void* const* d_in, const int* in_sizes, int n_in,
                              void* d_out, int out_size, void* d_ws,
                              size_t ws_size, hipStream_t stream) {
  const float* hid = (const float*)d_in[0];
  const float* Wq  = (const float*)d_in[1];
  const float* bq  = (const float*)d_in[2];
  const float* Wk  = (const float*)d_in[3];
  const float* bk  = (const float*)d_in[4];
  const float* Wv  = (const float*)d_in[5];
  const float* bv  = (const float*)d_in[6];
  const float* Wo  = (const float*)d_in[7];
  const float* bo  = (const float*)d_in[8];
  const float* gq  = (const float*)d_in[9];
  const float* gk  = (const float*)d_in[10];
  float* out = (float*)d_out;

  const size_t MB = 1ull << 20;
  char* ws = (char*)d_ws;
  _Float16* h16  = (_Float16*)(ws + 0 * MB);    // 16 MB  [L][HIDDEN]
  _Float16* wq16 = (_Float16*)(ws + 16 * MB);   // 8 MB   [N][K] transposed
  _Float16* wk16 = (_Float16*)(ws + 24 * MB);
  _Float16* wv16 = (_Float16*)(ws + 32 * MB);
  _Float16* wo16 = (_Float16*)(ws + 40 * MB);
  _Float16* q_t  = (_Float16*)(ws + 48 * MB);   // 16 MB [h][n][t][e]
  _Float16* k_t  = (_Float16*)(ws + 64 * MB);   // 16 MB [h][n][t][e]
  _Float16* v_t  = (_Float16*)(ws + 80 * MB);   // 16 MB [h][n][e][t]
  _Float16* o16  = (_Float16*)(ws + 96 * MB);   // 16 MB [L][HIDDEN]

  hcvt_kernel<<<(LSEQ * HIDDEN) / 256, 256, 0, stream>>>(hid, h16);
  wcvt_kernel<<<(HIDDEN * HIDDEN) / 256, 256, 0, stream>>>(Wq, wq16);
  wcvt_kernel<<<(HIDDEN * HIDDEN) / 256, 256, 0, stream>>>(Wk, wk16);
  wcvt_kernel<<<(HIDDEN * HIDDEN) / 256, 256, 0, stream>>>(Wv, wv16);
  wcvt_kernel<<<(HIDDEN * HIDDEN) / 256, 256, 0, stream>>>(Wo, wo16);

  qkv_kernel<<<dim3(LSEQ / 64, 3 * HEADS), 256, 0, stream>>>(
      h16, wq16, wk16, wv16, bq, bk, bv, gq, gk, q_t, k_t, v_t);

  attn_kernel<<<dim3(NTILES, HEADS, 2), 256, 0, stream>>>(q_t, k_t, v_t, o16);

  oproj_kernel<<<dim3(LSEQ / 64, HIDDEN / 128), 256, 0, stream>>>(o16, wo16, bo,
                                                                  out);
}
